// CenterBasedSeg_61856118997111
// MI455X (gfx1250) — compile-verified, matrix operand order checked by reading, not run on hardware
//
#include <hip/hip_runtime.h>
#include <hip/hip_bf16.h>
#include <math.h>

// ---------------- problem constants ----------------
#define N_PTS      250000
#define S_GAUSS    16
#define HIDDEN     256
#define L_LEV      12
#define FPL        2
#define TSIZE      (1 << 19)
#define ACTIVE     6
#define SHIFT_W    0.5f
#define DIM_PAD    96          // 91 padded to 3 * 32 (K tiles)
#define TILE       256         // points per block
#define HID_STRIDE 264         // 256 + 8 halves pad (avoid LDS bank aliasing)

typedef __attribute__((ext_vector_type(16))) _Float16 v16h;
typedef __attribute__((ext_vector_type(8)))  float    v8f;
typedef __attribute__((ext_vector_type(4)))  unsigned int v4u;

// ---------------- LDS layout (bytes) ----------------
#define OFF_W1T   0                                   // 256*96*2   = 49152
#define OFF_W2T   (OFF_W1T + 256*96*2)                // 32*264*2   = 16896
#define OFF_B1    (OFF_W2T + 32*HID_STRIDE*2)         // 256*4      = 1024
#define OFF_B2    (OFF_B1 + 256*4)                    // 32*4       = 128
#define OFF_DIST  (OFF_B2 + 128)                      // 256*16*4   = 16384
#define OFF_INFO  (OFF_DIST + TILE*S_GAUSS*4)         // 256*96*2   = 49152 (delta aliases)
#define OFF_HID   (OFF_INFO + TILE*DIM_PAD*2)         // 256*264*2  = 135168
#define SMEM_TOTAL (OFF_HID + TILE*HID_STRIDE*2)      // = 267904 bytes

// Load a 16-half WMMA fragment (A or B operand) from LDS, row-major matrix,
// per the CDNA5 16-bit operand layout: lane groups 0-15 / 16-31 take K offsets
// {0..7, 16..23} vs {8..15, 24..31} within the 32-wide K tile.
__device__ __forceinline__ v16h load_frag(const _Float16* m, int row, int stride, int kb) {
    v16h f;
    const _Float16* p = m + row * stride + kb;
    ((v4u*)&f)[0] = *(const v4u*)(p);        // K = kb .. kb+7     (16B -> ds_load_b128)
    ((v4u*)&f)[1] = *(const v4u*)(p + 16);   // K = kb+16 .. kb+23
    return f;
}

__device__ __forceinline__ v8f wmma_f16(v16h a, v16h b, v8f c) {
    return __builtin_amdgcn_wmma_f32_16x16x32_f16(false, a, false, b, (short)0, c, false, false);
}

__global__ __launch_bounds__(256, 1)
void CenterBasedSeg_61856118997111_kernel(
    const float* __restrict__ x,
    const float* __restrict__ tau_p,
    const float* __restrict__ center,
    const float* __restrict__ logscale,
    const float* __restrict__ rot,
    const float* __restrict__ W1,
    const float* __restrict__ b1,
    const float* __restrict__ W2,
    const float* __restrict__ b2,
    const float* __restrict__ tables,
    float* __restrict__ out)
{
    extern __shared__ __align__(16) char smem[];
    _Float16* W1T   = (_Float16*)(smem + OFF_W1T);   // [n=256][k=96]
    _Float16* W2T   = (_Float16*)(smem + OFF_W2T);   // [n=32][k=256] stride HID_STRIDE
    float*    b1s   = (float*)(smem + OFF_B1);
    float*    b2s   = (float*)(smem + OFF_B2);
    float*    distS = (float*)(smem + OFF_DIST);     // [256][16]
    _Float16* info  = (_Float16*)(smem + OFF_INFO);  // [256][96]
    float*    delta = (float*)(smem + OFF_INFO);     // aliases info (used after L1)
    _Float16* hid   = (_Float16*)(smem + OFF_HID);   // [256][HID_STRIDE]

    const int tid  = threadIdx.x;
    const int lane = tid & 31;
    const int wv   = tid >> 5;        // 8 waves
    const int rsel = (lane >> 4) & 1; // lane-half select
    const int colL = lane & 15;

    // ---------- stage 0: weights -> LDS (f16, transposed for B-operand) ----------
    for (int i = tid; i < HIDDEN * DIM_PAD; i += 256) {   // W1: [91][256] -> W1T[n][k]
        int k = i >> 8, n = i & 255;                       // coalesced read of W1 row k
        W1T[n * DIM_PAD + k] = (k < 91) ? (_Float16)W1[k * 256 + n] : (_Float16)0.f;
    }
    for (int i = tid; i < 256 * 32; i += 256) {           // W2: [256][32] -> W2T[n][k]
        int k = i >> 5, n = i & 31;
        W2T[n * HID_STRIDE + k] = (_Float16)W2[k * 32 + n];
    }
    if (tid < 256) b1s[tid] = b1[tid];
    if (tid < 32)  b2s[tid] = b2[tid];

    // ---------- stage 1: per-point features (1 thread = 1 point) ----------
    {
        const int p = blockIdx.x * TILE + tid;
        const bool valid = (p < N_PTS);
        float px = valid ? x[p * 3 + 0] : 0.f;
        float py = valid ? x[p * 3 + 1] : 0.f;
        float pz = valid ? x[p * 3 + 2] : 0.f;
        _Float16* row = info + tid * DIM_PAD;

        #pragma unroll 4
        for (int s = 0; s < S_GAUSS; ++s) {
            float qw = rot[s * 4 + 0], qx = rot[s * 4 + 1];
            float qy = rot[s * 4 + 2], qz = rot[s * 4 + 3];
            float rn = rsqrtf(qw * qw + qx * qx + qy * qy + qz * qz);
            qw *= rn; qx *= rn; qy *= rn; qz *= rn;
            float vx = px - center[s * 3 + 0];
            float vy = py - center[s * 3 + 1];
            float vz = pz - center[s * 3 + 2];
            // t = 2*cross(qv, v); r = v + qw*t + cross(qv, t)
            float tx = 2.f * (qy * vz - qz * vy);
            float ty = 2.f * (qz * vx - qx * vz);
            float tz = 2.f * (qx * vy - qy * vx);
            float rx = vx + qw * tx + (qy * tz - qz * ty);
            float ry = vy + qw * ty + (qz * tx - qx * tz);
            float rz = vz + qw * tz + (qx * ty - qy * tx);
            rx *= expf(-logscale[s * 3 + 0]);
            ry *= expf(-logscale[s * 3 + 1]);
            rz *= expf(-logscale[s * 3 + 2]);
            float d = rx * rx + ry * ry + rz * rz;
            row[s * 4 + 0] = (_Float16)rx;
            row[s * 4 + 1] = (_Float16)ry;
            row[s * 4 + 2] = (_Float16)rz;
            row[s * 4 + 3] = (_Float16)sqrtf(d);
            distS[tid * S_GAUSS + s] = d;
        }

        // hash grid: 6 active levels, 8 trilinear corners each
        const float res[ACTIVE] = {16.f, 24.f, 36.f, 54.f, 81.f, 121.f};
        for (int l = 0; l < ACTIVE; ++l) {
            float fx = px * res[l], fy = py * res[l], fz = pz * res[l];
            float gx = floorf(fx), gy = floorf(fy), gz = floorf(fz);
            float wx = fx - gx, wy = fy - gy, wz = fz - gz;
            int ix = (int)gx, iy = (int)gy, iz = (int)gz;
            const float* tb = tables + (size_t)l * TSIZE * FPL;
            float a0 = 0.f, a1 = 0.f;
            #pragma unroll
            for (int c = 0; c < 8; ++c) {
                int ox = (c >> 2) & 1, oy = (c >> 1) & 1, oz = c & 1;
                unsigned int h = (unsigned int)(ix + ox) * 1u
                               ^ (unsigned int)(iy + oy) * 2654435761u
                               ^ (unsigned int)(iz + oz) * 805459861u;
                unsigned int idx = h & (unsigned int)(TSIZE - 1);
                const float* f = tb + (size_t)idx * FPL;
                float wt = (ox ? wx : 1.f - wx) * (oy ? wy : 1.f - wy) * (oz ? wz : 1.f - wz);
                a0 = fmaf(wt, f[0], a0);
                a1 = fmaf(wt, f[1], a1);
            }
            row[64 + l * 2 + 0] = (_Float16)a0;
            row[64 + l * 2 + 1] = (_Float16)a1;
        }
        for (int l = ACTIVE; l < L_LEV; ++l) {            // inactive levels -> 0
            row[64 + l * 2 + 0] = (_Float16)0.f;
            row[64 + l * 2 + 1] = (_Float16)0.f;
        }
        row[88] = (_Float16)px; row[89] = (_Float16)py; row[90] = (_Float16)pz;
        row[91] = (_Float16)0.f; row[92] = (_Float16)0.f;
        row[93] = (_Float16)0.f; row[94] = (_Float16)0.f; row[95] = (_Float16)0.f;
    }
    __syncthreads();

    // ---------- stage 2: layer 1 = relu(info @ W1 + b1)  (WMMA f16->f32) ----------
    // wave wv owns M-tiles {wv, wv+8}; 16 N-tiles; K = 3 tiles of 32
    {
        v16h a[2][3];
        #pragma unroll
        for (int ms = 0; ms < 2; ++ms) {
            int mt = wv + ms * 8;
            #pragma unroll
            for (int kt = 0; kt < 3; ++kt)
                a[ms][kt] = load_frag(info, mt * 16 + colL, DIM_PAD, kt * 32 + rsel * 8);
        }
        for (int nt = 0; nt < 16; ++nt) {
            v16h b0 = load_frag(W1T, nt * 16 + colL, DIM_PAD, 0 * 32 + rsel * 8);
            v16h b1f = load_frag(W1T, nt * 16 + colL, DIM_PAD, 1 * 32 + rsel * 8);
            v16h b2f = load_frag(W1T, nt * 16 + colL, DIM_PAD, 2 * 32 + rsel * 8);
            #pragma unroll
            for (int ms = 0; ms < 2; ++ms) {
                v8f acc = {};
                acc = wmma_f16(a[ms][0], b0, acc);
                acc = wmma_f16(a[ms][1], b1f, acc);
                acc = wmma_f16(a[ms][2], b2f, acc);
                int col = nt * 16 + colL;
                float bb = b1s[col];
                int mt = wv + ms * 8;
                #pragma unroll
                for (int r = 0; r < 8; ++r) {
                    int rw = mt * 16 + rsel * 8 + r;
                    float h = acc[r] + bb;
                    hid[rw * HID_STRIDE + col] = (_Float16)(h > 0.f ? h : 0.f);
                }
            }
        }
    }
    __syncthreads();

    // ---------- stage 3: layer 2 = hidden @ W2 + b2 ----------
    {
        #pragma unroll
        for (int ms = 0; ms < 2; ++ms) {
            int mt = wv + ms * 8;
            v16h a[8];
            #pragma unroll
            for (int kt = 0; kt < 8; ++kt)
                a[kt] = load_frag(hid, mt * 16 + colL, HID_STRIDE, kt * 32 + rsel * 8);
            #pragma unroll
            for (int nt = 0; nt < 2; ++nt) {
                v8f acc = {};
                #pragma unroll
                for (int kt = 0; kt < 8; ++kt) {
                    v16h b = load_frag(W2T, nt * 16 + colL, HID_STRIDE, kt * 32 + rsel * 8);
                    acc = wmma_f16(a[kt], b, acc);
                }
                float bb = b2s[nt * 16 + colL];
                #pragma unroll
                for (int r = 0; r < 8; ++r) {
                    int rw = mt * 16 + rsel * 8 + r;
                    delta[rw * 32 + nt * 16 + colL] = acc[r] + bb;
                }
            }
        }
    }
    __syncthreads();

    // ---------- stage 4: softmax epilogue (1 thread = 1 point) ----------
    {
        const int p = blockIdx.x * TILE + tid;
        if (p < N_PTS) {
            const float invT = (float)(S_GAUSS - 1) / tau_p[0];
            float z[S_GAUSS];
            float m = -3.4e38f;
            #pragma unroll
            for (int s = 0; s < S_GAUSS; ++s) {
                float ls = delta[tid * 32 + s];
                float sh = delta[tid * 32 + 16 + s];
                float d  = distS[tid * S_GAUSS + s] * expf(SHIFT_W * ls);
                float lg = (-d + sh * SHIFT_W) * invT;
                z[s] = lg;
                m = fmaxf(m, lg);
            }
            float sum = 0.f;
            #pragma unroll
            for (int s = 0; s < S_GAUSS; ++s) { z[s] = expf(z[s] - m); sum += z[s]; }
            float inv = 1.f / sum;
            #pragma unroll
            for (int s = 0; s < S_GAUSS; ++s) out[(size_t)p * S_GAUSS + s] = z[s] * inv;
        }
    }
}

extern "C" void kernel_launch(void* const* d_in, const int* in_sizes, int n_in,
                              void* d_out, int out_size, void* d_ws, size_t ws_size,
                              hipStream_t stream) {
    const float* x        = (const float*)d_in[0];
    const float* tau      = (const float*)d_in[1];
    const float* center   = (const float*)d_in[2];
    const float* logscale = (const float*)d_in[3];
    const float* rot      = (const float*)d_in[4];
    const float* W1       = (const float*)d_in[5];
    const float* b1       = (const float*)d_in[6];
    const float* W2       = (const float*)d_in[7];
    const float* b2       = (const float*)d_in[8];
    const float* tables   = (const float*)d_in[9];
    float* out            = (float*)d_out;

    const int grid = (N_PTS + TILE - 1) / TILE;   // 977 blocks
    hipLaunchKernelGGL(CenterBasedSeg_61856118997111_kernel,
                       dim3(grid), dim3(256), SMEM_TOTAL, stream,
                       x, tau, center, logscale, rot, W1, b1, W2, b2, tables, out);
}
